// W_GATCR_CL_90005334655075
// MI455X (gfx1250) — matrix operand order repaired and checked by reading, not compiled, not run on hardware
//
#include <hip/hip_runtime.h>
#include <math.h>

typedef __attribute__((ext_vector_type(16))) __bf16 v16bf;
typedef __attribute__((ext_vector_type(8)))  __bf16 v8bf;
typedef __attribute__((ext_vector_type(8)))  float  v8f;

#define LN_EPS_F 1e-5f
#define NEG_SLOPE_F 0.2f

// ---------------- fp32 -> bf16 conversion (row-major) ----------------
__global__ void k_cvt_bf16(const float* __restrict__ in, __bf16* __restrict__ out, long n) {
  long i = blockIdx.x * (long)blockDim.x + threadIdx.x;
  if (i < n) out[i] = (__bf16)in[i];
}

// ---------------- fp32 -> bf16 with transpose: out[c*R + r] = in[r*C + c] ----------------
__global__ void k_cvt_bf16_T(const float* __restrict__ in, __bf16* __restrict__ out,
                             int R, int Cc) {
  int i = blockIdx.x * blockDim.x + threadIdx.x;
  if (i >= R * Cc) return;
  const int r = i / Cc, c = i - r * Cc;
  out[(long)c * R + r] = (__bf16)in[i];
}

// ---------------- GEMM via WMMA (bf16 in, f32 accumulate) ----------------
// C[N x Ncols] = Xb[N x K] * W, Wt = W^T in bf16 [Ncols x K], staged in LDS with
// CDNA5 async-to-LDS loads. Block = 256 threads (8 waves); wave owns 16 rows and
// sweeps all column tiles with its 4 A fragments held in VGPRs.
__global__ void __launch_bounds__(256)
k_gemm_wmma(const __bf16* __restrict__ Xb, const __bf16* __restrict__ Wt,
            float* __restrict__ C, int N, int K, int Ncols) {
  __shared__ __align__(16) __bf16 sB[128 * 128];   // up to 32KB
  const int tid  = threadIdx.x;
  const int wave = tid >> 5;
  const int lane = tid & 31;
  const int half = lane >> 4;
  const int l16  = lane & 15;

  // ---- stage Wt (Ncols*K bf16) into LDS via async tensor-path loads ----
  {
    const int nchunks = (Ncols * K * 2) >> 4;       // 16-byte chunks
    const char* gbase = (const char*)Wt;
    char* lbase = (char*)&sB[0];
    for (int i = tid; i < nchunks; i += 256) {
      unsigned lds = (unsigned)(size_t)(lbase + i * 16);
      unsigned long long ga = (unsigned long long)(size_t)(gbase + i * 16);
      asm volatile("global_load_async_to_lds_b128 %0, %1, off"
                   :: "v"(lds), "v"(ga) : "memory");
    }
    asm volatile("s_wait_asynccnt 0" ::: "memory");
    __syncthreads();
  }

  const int rowTile = blockIdx.x * 128 + wave * 16;
  int m = rowTile + l16;
  if (m >= N) m = N - 1;                            // clamp: EXEC stays all-ones for WMMA
  __builtin_prefetch(&Xb[(long)m * K], 0, 0);

  v16bf afrag[4];
#pragma unroll
  for (int ks = 0; ks < 4; ++ks) {
    const int kbA = ks * 32 + half * 8;
    const v8bf lo = *(const v8bf*)(Xb + (long)m * K + kbA);
    const v8bf hi = *(const v8bf*)(Xb + (long)m * K + kbA + 16);
    afrag[ks] = __builtin_shufflevector(lo, hi, 0, 1, 2, 3, 4, 5, 6, 7,
                                        8, 9, 10, 11, 12, 13, 14, 15);
  }

  const bool fullTile = (rowTile + 16 <= N);        // uniform per wave
  const int nColTiles = Ncols >> 4;
  for (int j = 0; j < nColTiles; ++j) {
    const int n = j * 16 + l16;
    v8f acc = {};
#pragma unroll
    for (int ks = 0; ks < 4; ++ks) {
      const int kbB = ks * 32 + half * 8;
      const v8bf lo = *(const v8bf*)(&sB[n * K + kbB]);
      const v8bf hi = *(const v8bf*)(&sB[n * K + kbB + 16]);
      const v16bf b = __builtin_shufflevector(lo, hi, 0, 1, 2, 3, 4, 5, 6, 7,
                                              8, 9, 10, 11, 12, 13, 14, 15);
      acc = __builtin_amdgcn_wmma_f32_16x16x32_bf16(false, afrag[ks], false, b,
                                                    (short)0, acc, false, false);
    }
    if (fullTile) {
#pragma unroll
      for (int r = 0; r < 8; ++r)
        C[(long)(rowTile + half * 8 + r) * Ncols + j * 16 + l16] = acc[r];
    } else {
#pragma unroll
      for (int r = 0; r < 8; ++r) {
        const int mm = rowTile + half * 8 + r;
        if (mm < N) C[(long)mm * Ncols + j * 16 + l16] = acc[r];
      }
    }
  }
}

// ---------------- per-node attention logits ----------------
__global__ void k_el_er(const float* __restrict__ ft, const float* __restrict__ al,
                        const float* __restrict__ ar, float* __restrict__ el,
                        float* __restrict__ er, int N, int F, int D) {
  const int node = blockIdx.x * (blockDim.x >> 5) + (threadIdx.x >> 5);
  const int lane = threadIdx.x & 31;
  if (node >= N) return;
  const int VPT = F >> 5;
  const int G = D / VPT;
  float sl = 0.f, sr = 0.f;
  for (int j = 0; j < VPT; ++j) {
    const int f = lane * VPT + j;
    const float v = ft[(long)node * F + f];
    sl += v * al[f];
    sr += v * ar[f];
  }
  for (int off = G >> 1; off > 0; off >>= 1) {
    sl += __shfl_xor(sl, off);
    sr += __shfl_xor(sr, off);
  }
  if ((lane % G) == 0) {
    const int H = F / D;
    const int h = lane / G;
    el[(long)node * H + h] = sl;
    er[(long)node * H + h] = sr;
  }
}

// ================= CSR build (dst is layer-invariant: built once) =================
__global__ void k_zero_int(int* __restrict__ p, int n) {
  int i = blockIdx.x * blockDim.x + threadIdx.x;
  if (i < n) p[i] = 0;
}
__global__ void k_count_dst(const int* __restrict__ dst, int* __restrict__ counts, int E) {
  int i = blockIdx.x * blockDim.x + threadIdx.x;
  if (i < E) atomicAdd(&counts[dst[i]], 1);
}
// single-block exclusive scan (N up to a few hundred K)
__global__ void __launch_bounds__(1024) k_scan_excl(const int* __restrict__ counts,
                                                    int* __restrict__ offs, int N) {
  __shared__ int sdata[1024];
  __shared__ int s_running;
  const int t = threadIdx.x;
  if (t == 0) s_running = 0;
  __syncthreads();
  for (int base = 0; base < N; base += 1024) {
    const int v = (base + t < N) ? counts[base + t] : 0;
    sdata[t] = v;
    __syncthreads();
    for (int off = 1; off < 1024; off <<= 1) {
      const int add = (t >= off) ? sdata[t - off] : 0;
      __syncthreads();
      sdata[t] += add;
      __syncthreads();
    }
    const int incl = sdata[t];
    const int run = s_running;
    if (base + t < N) offs[base + t] = run + incl - v;
    __syncthreads();
    if (t == 1023) s_running = run + incl;
    __syncthreads();
  }
  if (t == 0) offs[N] = s_running;
}
__global__ void k_copy_int(const int* __restrict__ a, int* __restrict__ b, int n) {
  int i = blockIdx.x * blockDim.x + threadIdx.x;
  if (i < n) b[i] = a[i];
}
__global__ void k_scatter_dst(const int* __restrict__ dst, int* __restrict__ cursor,
                              int* __restrict__ eidx, int E) {
  int i = blockIdx.x * blockDim.x + threadIdx.x;
  if (i < E) eidx[atomicAdd(&cursor[dst[i]], 1)] = i;
}
// canonicalize per-node edge order (insertion sort; avg degree ~16) -> deterministic fp sums
__global__ void k_sort_seg(const int* __restrict__ offs, int* __restrict__ eidx, int N) {
  int n = blockIdx.x * blockDim.x + threadIdx.x;
  if (n >= N) return;
  const int beg = offs[n], end = offs[n + 1];
  for (int i = beg + 1; i < end; ++i) {
    const int key = eidx[i];
    int j = i - 1;
    while (j >= beg && eidx[j] > key) { eidx[j + 1] = eidx[j]; --j; }
    eidx[j + 1] = key;
  }
}

// ================= node-centric edge softmax: a[p,h] written per CSR slot =================
__global__ void __launch_bounds__(128)
k_node_softmax(const int* __restrict__ offs, const int* __restrict__ eidx,
               const int* __restrict__ srcArr, const float* __restrict__ ew,
               const float* __restrict__ el, const float* __restrict__ er,
               const float* __restrict__ lam, float* __restrict__ abuf, int N, int H) {
  const int node = blockIdx.x * (blockDim.x >> 5) + (threadIdx.x >> 5);
  const int lane = threadIdx.x & 31;
  if (node >= N) return;
  const int beg = offs[node], end = offs[node + 1];
  const int deg = end - beg;
  const float lamv = lam[0];

  if (deg <= 32) {                       // fast path: edge data cached in registers
    int srcn = 0; float ewv = 0.f;
    const bool act = (lane < deg);
    if (act) { const int e = eidx[beg + lane]; srcn = srcArr[e]; ewv = ew[e]; }
    for (int h = 0; h < H; ++h) {
      const float erv = er[(long)node * H + h];
      float v = -1e30f;
      if (act) {
        v = el[(long)srcn * H + h] + erv + lamv * ewv;
        v = (v > 0.f) ? v : NEG_SLOPE_F * v;
      }
      float mx = v;
      for (int off = 16; off > 0; off >>= 1) mx = fmaxf(mx, __shfl_xor(mx, off));
      const float ex = act ? __expf(v - mx) : 0.f;
      float sum = ex;
      for (int off = 16; off > 0; off >>= 1) sum += __shfl_xor(sum, off);
      if (act) abuf[(long)(beg + lane) * H + h] = ex / sum;
    }
  } else {                               // general path
    for (int h = 0; h < H; ++h) {
      const float erv = er[(long)node * H + h];
      float mx = -1e30f;
      for (int p = beg + lane; p < end; p += 32) {
        const int e = eidx[p];
        float v = el[(long)srcArr[e] * H + h] + erv + lamv * ew[e];
        v = (v > 0.f) ? v : NEG_SLOPE_F * v;
        mx = fmaxf(mx, v);
      }
      for (int off = 16; off > 0; off >>= 1) mx = fmaxf(mx, __shfl_xor(mx, off));
      float sum = 0.f;
      for (int p = beg + lane; p < end; p += 32) {
        const int e = eidx[p];
        float v = el[(long)srcArr[e] * H + h] + erv + lamv * ew[e];
        v = (v > 0.f) ? v : NEG_SLOPE_F * v;
        sum += __expf(v - mx);
      }
      for (int off = 16; off > 0; off >>= 1) sum += __shfl_xor(sum, off);
      const float inv = 1.f / sum;
      for (int p = beg + lane; p < end; p += 32) {
        const int e = eidx[p];
        float v = el[(long)srcArr[e] * H + h] + erv + lamv * ew[e];
        v = (v > 0.f) ? v : NEG_SLOPE_F * v;
        abuf[(long)p * H + h] = __expf(v - mx) * inv;
      }
    }
  }
}

// ================= node-centric gather aggregation (no atomics) =================
__global__ void __launch_bounds__(128)
k_node_aggregate_f128(const int* __restrict__ offs, const int* __restrict__ eidx,
                      const int* __restrict__ srcArr, const float* __restrict__ abuf,
                      const float* __restrict__ ft, float* __restrict__ agg, int N) {
  const int node = blockIdx.x * (blockDim.x >> 5) + (threadIdx.x >> 5);
  const int lane = threadIdx.x & 31;
  if (node >= N) return;
  const int beg = offs[node], end = offs[node + 1];
  const int hidx = lane >> 3;            // F=128, H=4, D=32: head = (lane*4)/32
  float a0 = 0.f, a1 = 0.f, a2 = 0.f, a3 = 0.f;
  for (int p = beg; p < end; ++p) {
    const int s = srcArr[eidx[p]];
    const float a = abuf[(long)p * 4 + hidx];
    const float4 rv = *(const float4*)(ft + (long)s * 128 + lane * 4);
    a0 += rv.x * a; a1 += rv.y * a; a2 += rv.z * a; a3 += rv.w * a;
  }
  *(float4*)(agg + (long)node * 128 + lane * 4) = make_float4(a0, a1, a2, a3);
}

__global__ void __launch_bounds__(128)
k_node_aggregate_f64(const int* __restrict__ offs, const int* __restrict__ eidx,
                     const int* __restrict__ srcArr, const float* __restrict__ abuf,
                     const float* __restrict__ ft, float* __restrict__ agg, int N) {
  const int node = blockIdx.x * (blockDim.x >> 5) + (threadIdx.x >> 5);
  const int lane = threadIdx.x & 31;
  if (node >= N) return;
  const int beg = offs[node], end = offs[node + 1];
  float a0 = 0.f, a1 = 0.f;
  for (int p = beg; p < end; ++p) {      // H=1
    const int s = srcArr[eidx[p]];
    const float a = abuf[p];
    const float2 rv = *(const float2*)(ft + (long)s * 64 + lane * 2);
    a0 += rv.x * a; a1 += rv.y * a;
  }
  *(float2*)(agg + (long)node * 64 + lane * 2) = make_float2(a0, a1);
}

// ---------------- ELU + LayerNorm + residual (F=128), one wave per node ----------------
__global__ void k_post_elu_ln_res(const float* __restrict__ agg, const float* __restrict__ res,
                                  const float* __restrict__ g, const float* __restrict__ b,
                                  float* __restrict__ out, int N) {
  const int node = blockIdx.x * (blockDim.x >> 5) + (threadIdx.x >> 5);
  const int lane = threadIdx.x & 31;
  if (node >= N) return;
  float x[4];
  float s = 0.f;
#pragma unroll
  for (int j = 0; j < 4; ++j) {
    const int f = lane * 4 + j;
    float v = agg[(long)node * 128 + f];
    v = (v > 0.f) ? v : (__expf(v) - 1.f);   // ELU(alpha=1)
    x[j] = v;
    s += v;
  }
  for (int off = 16; off > 0; off >>= 1) s += __shfl_xor(s, off);
  const float mu = s * (1.f / 128.f);
  float var = 0.f;
#pragma unroll
  for (int j = 0; j < 4; ++j) { const float d = x[j] - mu; var += d * d; }
  for (int off = 16; off > 0; off >>= 1) var += __shfl_xor(var, off);
  const float rstd = rsqrtf(var * (1.f / 128.f) + LN_EPS_F);
#pragma unroll
  for (int j = 0; j < 4; ++j) {
    const int f = lane * 4 + j;
    out[(long)node * 128 + f] = (x[j] - mu) * rstd * g[f] + b[f] + res[(long)node * 128 + f];
  }
}

// ---------------- final: duplicate output (tuple of identical tensors) ----------------
__global__ void k_copy_out2(const float* __restrict__ agg, float* __restrict__ out, int n) {
  int i = blockIdx.x * blockDim.x + threadIdx.x;
  if (i < n) { const float v = agg[i]; out[i] = v; out[n + i] = v; }
}

extern "C" void kernel_launch(void* const* d_in, const int* in_sizes, int n_in,
                              void* d_out, int out_size, void* d_ws, size_t ws_size,
                              hipStream_t stream) {
  (void)n_in; (void)out_size; (void)ws_size;
  const float* features = (const float*)d_in[0];
  const float* ew   = (const float*)d_in[1];
  const int*   src  = (const int*)d_in[2];
  const int*   dst  = (const int*)d_in[3];
  const float* W0   = (const float*)d_in[4];
  const float* al0  = (const float*)d_in[5];
  const float* ar0  = (const float*)d_in[6];
  const float* lam0 = (const float*)d_in[7];
  const float* W1   = (const float*)d_in[8];
  const float* al1  = (const float*)d_in[9];
  const float* ar1  = (const float*)d_in[10];
  const float* lam1 = (const float*)d_in[11];
  const float* W2   = (const float*)d_in[12];
  const float* al2  = (const float*)d_in[13];
  const float* ar2  = (const float*)d_in[14];
  const float* lam2 = (const float*)d_in[15];
  const float* g0   = (const float*)d_in[16];
  const float* b0   = (const float*)d_in[17];
  const float* g1   = (const float*)d_in[18];
  const float* b1   = (const float*)d_in[19];

  const int INF_ = 128, HD = 128, OUTF = 64, H = 4, D = 32;
  const int N = in_sizes[0] / INF_;
  const int E = in_sizes[1];

  char* w = (char*)d_ws;
  auto carve = [&](size_t bytes) {
    void* p = (void*)w;
    w += (bytes + 255) & ~(size_t)255;
    return p;
  };
  float*  ft     = (float*)carve((size_t)N * HD * 4);
  float*  h1     = (float*)carve((size_t)N * HD * 4);
  float*  h2     = (float*)carve((size_t)N * HD * 4);
  float*  agg    = (float*)carve((size_t)N * HD * 4);
  float*  el     = (float*)carve((size_t)N * H * 4);
  float*  er     = (float*)carve((size_t)N * H * 4);
  float*  abuf   = (float*)carve((size_t)E * H * 4);
  int*    counts = (int*)carve((size_t)N * 4);
  int*    cursor = (int*)carve((size_t)N * 4);
  int*    offs   = (int*)carve((size_t)(N + 1) * 4);
  int*    eidx   = (int*)carve((size_t)E * 4);
  __bf16* Xb     = (__bf16*)carve((size_t)N * HD * 2);
  __bf16* Wb     = (__bf16*)carve((size_t)HD * HD * 2);

  // ---- CSR build over dst (graph is identical for all 3 layers) ----
  k_zero_int<<<(N + 255) / 256, 256, 0, stream>>>(counts, N);
  k_count_dst<<<(E + 255) / 256, 256, 0, stream>>>(dst, counts, E);
  k_scan_excl<<<1, 1024, 0, stream>>>(counts, offs, N);
  k_copy_int<<<(N + 255) / 256, 256, 0, stream>>>(offs, cursor, N);
  k_scatter_dst<<<(E + 255) / 256, 256, 0, stream>>>(dst, cursor, eidx, E);
  k_sort_seg<<<(N + 255) / 256, 256, 0, stream>>>(offs, eidx, N);

  auto gat = [&](const float* X, const float* Wm, const float* alp, const float* arp,
                 const float* lamp, int Fin, int Fout, int Hh, int Dd, float* aggOut) {
    const long nX = (long)N * Fin;
    k_cvt_bf16<<<(unsigned)((nX + 255) / 256), 256, 0, stream>>>(X, Xb, nX);
    k_cvt_bf16_T<<<((Fin * Fout) + 255) / 256, 256, 0, stream>>>(Wm, Wb, Fin, Fout);
    k_gemm_wmma<<<(N + 127) / 128, 256, 0, stream>>>(Xb, Wb, ft, N, Fin, Fout);
    k_el_er<<<(N + 3) / 4, 128, 0, stream>>>(ft, alp, arp, el, er, N, Fout, Dd);
    k_node_softmax<<<(N + 3) / 4, 128, 0, stream>>>(offs, eidx, src, ew, el, er, lamp,
                                                    abuf, N, Hh);
    if (Fout == 128)
      k_node_aggregate_f128<<<(N + 3) / 4, 128, 0, stream>>>(offs, eidx, src, abuf,
                                                             ft, aggOut, N);
    else
      k_node_aggregate_f64<<<(N + 3) / 4, 128, 0, stream>>>(offs, eidx, src, abuf,
                                                            ft, aggOut, N);
  };

  // layer 0: GAT -> ELU -> LN -> +features
  gat(features, W0, al0, ar0, lam0, INF_, HD, H, D, agg);
  k_post_elu_ln_res<<<(N + 3) / 4, 128, 0, stream>>>(agg, features, g0, b0, h1, N);
  // layer 1: GAT -> ELU -> LN -> +h1
  gat(h1, W1, al1, ar1, lam1, HD, HD, H, D, agg);
  k_post_elu_ln_res<<<(N + 3) / 4, 128, 0, stream>>>(agg, h1, g1, b1, h2, N);
  // layer 2: GAT (single head, no activation), output duplicated (tuple)
  gat(h2, W2, al2, ar2, lam2, HD, OUTF, 1, OUTF, agg);
  k_copy_out2<<<((N * OUTF) + 255) / 256, 256, 0, stream>>>(agg, (float*)d_out, N * OUTF);
}
